// Decoder_41437844472390
// MI455X (gfx1250) — matrix operand order repaired and checked by reading, not compiled, not run on hardware
//
#include <hip/hip_runtime.h>
#include <hip/hip_bf16.h>
#include <math.h>

// ---- problem constants (from reference) ----
#define B_    32
#define T_    128
#define V_    32000
#define E_    300
#define H_    512
#define FOURH 2048
#define KP_E  320            // E padded to multiple of 32 (WMMA K step)
#define MROWS 4096           // B*T

typedef __attribute__((ext_vector_type(16))) __bf16    v16bf;
typedef __attribute__((ext_vector_type(8)))  float     v8f;
typedef __attribute__((ext_vector_type(4)))  unsigned  v4u_t;
typedef __attribute__((ext_vector_type(8)))  int       v8i_t;
typedef __attribute__((ext_vector_type(4)))  int       v4i_t;

// ---------------------------------------------------------------------------
// WMMA fragment loaders (bf16, wave32). Layouts used here (contiguous K, same order in A and B):
//  A (16xK tile, MxK): lane L<16 -> row M=L holds K = k0+0..15 contiguous;
//                      lane L>=16 -> row M=L-16 holds K = k0+16..31.
//  B (Kx16 tile, stored as BT[N][K] row-major): lane n<16 -> col n holds
//                      K = k0+0..15 contiguous; lane n+16 -> K = k0+16..31.
// Generic pointers so the same helpers serve global and LDS tiles.
// ---------------------------------------------------------------------------
__device__ __forceinline__ v16bf load_fragA(const __bf16* base, int ld,
                                            int row0, int k0, int lane) {
  const __bf16* p = base + (size_t)(row0 + (lane & 15)) * ld + k0 + ((lane >> 4) << 4);
  v16bf r;
  float4* rp = reinterpret_cast<float4*>(&r);
  rp[0] = *reinterpret_cast<const float4*>(p);        // K = base..base+7
  rp[1] = *reinterpret_cast<const float4*>(p + 8);    // K = base+8..base+15
  return r;
}

__device__ __forceinline__ v16bf load_fragB(const __bf16* baseT, int ld,
                                            int col0, int k0, int lane) {
  const __bf16* p = baseT + (size_t)(col0 + (lane & 15)) * ld + k0 + ((lane >> 4) << 4);
  v16bf r;
  float4* rp = reinterpret_cast<float4*>(&r);
  rp[0] = *reinterpret_cast<const float4*>(p);        // 8 contiguous K
  rp[1] = *reinterpret_cast<const float4*>(p + 8);    // next 8 contiguous K
  return r;
}

__device__ __forceinline__ float sigmoidf_(float x) { return 1.0f / (1.0f + __expf(-x)); }

// ---------------------------------------------------------------------------
// gfx1250 async global->LDS copy (ASYNCcnt) + TDM tensor load (TENSORcnt).
// LDS destination address = low 32 bits of the generic pointer (aperture rule).
// ---------------------------------------------------------------------------
__device__ __forceinline__ unsigned lds_addr_of(const void* p) {
  return (unsigned)(uintptr_t)p;
}

__device__ __forceinline__ void async_copy_b128(unsigned ldsAddr, unsigned gByteOff,
                                                const void* base) {
  asm volatile("global_load_async_to_lds_b128 %0, %1, %2"
               :: "v"(ldsAddr), "v"(gByteOff), "s"(base)
               : "memory");
}

__device__ __forceinline__ void wait_async_all() {
  asm volatile("s_wait_asynccnt 0x0" ::: "memory");
}

// TDM: DMA a 2D tile of 256 rows x 64 bytes (row pitch = K*2 bytes) into LDS.
// D# per CDNA5 ISA ch.8: data_size=3 (8-byte units) -> tile_dim0 = 8 units,
// tensor_dim0 = tensor_dim0_stride = K/4 units, tile_dim1 = tensor_dim1 = 256.
// This toolchain's builtin takes 6 args: (g0, g1, g2, g3, extra v8i, cpol).
__device__ __forceinline__ void tdm_load_b_tile(unsigned ldsAddr,
                                                unsigned long long gaddr, int K) {
  const unsigned td0 = ((unsigned)K) >> 2;            // K*2 bytes / 8-byte units
  v4u_t g0;
  g0[0] = 1u;                                          // count=1, user descriptor
  g0[1] = ldsAddr;                                     // lds_addr [63:32]
  g0[2] = (unsigned)gaddr;                             // global_addr lo
  g0[3] = ((unsigned)(gaddr >> 32) & 0x01FFFFFFu) | (2u << 30); // addr hi | type=2
  v8i_t g1 = {};
  g1[0] = (int)(3u << 16);                             // wg_mask=0 | data_size=3 (8B)
  g1[1] = (int)((td0 & 0xFFFFu) << 16);                // tensor_dim0 lo16
  g1[2] = (int)(((td0 >> 16) & 0xFFFFu) | (256u << 16)); // tensor_dim0 hi | tensor_dim1 lo
  g1[3] = (int)(8u << 16);                             // tensor_dim1 hi=0 | tile_dim0=8
  g1[4] = (int)256u;                                   // tile_dim1=256 | tile_dim2=0
  g1[5] = (int)td0;                                    // tensor_dim0_stride lo32
  g1[6] = 0;                                           // stride hi | dim1_stride lo
  g1[7] = 0;
  v4i_t gz4 = {};                                      // groups 2/3: zero (2D tile)
  v8i_t gz8 = {};
  __builtin_amdgcn_tensor_load_to_lds(g0, g1, gz4, gz4, gz8, 0);
}

__device__ __forceinline__ void wait_tensor_all() {
  __builtin_amdgcn_s_wait_tensorcnt(0);
}

// ---------------------------------------------------------------------------
// Tiled transpose + f32->bf16 convert:  W[K][N] (row-major f32)  ->
// WT[N][Kp] (row-major bf16), zero-padding K..Kp.
// ---------------------------------------------------------------------------
__global__ void transpose_convert_kernel(const float* __restrict__ W, __bf16* __restrict__ WT,
                                         int K, int N, int Kp) {
  __shared__ float tile[32][33];
  const int nt = blockIdx.x * 32;
  const int kt = blockIdx.y * 32;
#pragma unroll
  for (int i = 0; i < 4; ++i) {
    int k = kt + threadIdx.y + i * 8;
    int n = nt + threadIdx.x;
    tile[threadIdx.y + i * 8][threadIdx.x] = (k < K && n < N) ? W[(size_t)k * N + n] : 0.0f;
  }
  __syncthreads();
#pragma unroll
  for (int i = 0; i < 4; ++i) {
    int n = nt + threadIdx.y + i * 8;
    int k = kt + threadIdx.x;
    if (n < N && k < Kp)
      WT[(size_t)n * Kp + k] = (__bf16)tile[threadIdx.x][threadIdx.y + i * 8];
  }
}

// ---------------------------------------------------------------------------
// Embedding gather -> padded bf16 A matrix  x[MROWS][KP_E]
// ---------------------------------------------------------------------------
__global__ void embed_kernel(const int* __restrict__ inputs, const float* __restrict__ emb,
                             __bf16* __restrict__ x) {
  int idx = blockIdx.x * blockDim.x + threadIdx.x;
  if (idx >= MROWS * KP_E) return;
  int m = idx / KP_E, k = idx - m * KP_E;
  float v = (k < E_) ? emb[(size_t)inputs[m] * E_ + k] : 0.0f;
  x[idx] = (__bf16)v;
}

__global__ void init_state_kernel(const float* __restrict__ h0, const float* __restrict__ c0,
                                  float* __restrict__ h, float* __restrict__ c,
                                  __bf16* __restrict__ hb0) {
  int i = blockIdx.x * blockDim.x + threadIdx.x;
  if (i >= B_ * H_) return;
  h[i] = h0[i];
  c[i] = c0[i];
  hb0[i] = (__bf16)h0[i];
}

// ---------------------------------------------------------------------------
// bf16 WMMA GEMM, double-buffered LDS staging:
//   C[M][ldc](f32) = A[M][K](bf16) * BT[N][K]^T + bias[N]
// Block = 256 threads = 8 waves (2 waveM x 4 waveN); macro tile 64x256, BK=32.
//  - B tile (256x32 bf16 = 16 KB): ONE tensor_load_to_lds DMA per K-step,
//    issued by wave 0, tracked by TENSORcnt.
//  - A tile (64x32 bf16 = 4 KB): per-lane global_load_async_to_lds_b128
//    (ASYNCcnt), one b128 per thread.
// Compute reads fragments via ds_load_b128 and issues 8 v_wmma per step/wave.
// Requires M % 64 == 0, N % 256 == 0, K % 32 == 0 (true for both call sites).
// ---------------------------------------------------------------------------
__launch_bounds__(256)
__global__ void gemm_bf16_wmma_kernel(const __bf16* __restrict__ A, const __bf16* __restrict__ BT,
                                      const float* __restrict__ bias, float* __restrict__ C,
                                      int K, int ldc) {
  __shared__ __align__(16) __bf16 As[2][64][32];    // 2 x 4 KB  (stride 1<<12)
  __shared__ __align__(16) __bf16 Bs[2][256][32];   // 2 x 16 KB (stride 1<<14)

  const int tid   = threadIdx.x;
  const int lane  = tid & 31;
  const int wave  = tid >> 5;
  const int waveM = wave >> 2;                 // 0..1
  const int waveN = wave & 3;                  // 0..3
  const int mBlock = blockIdx.y * 64;
  const int nBlock = blockIdx.x * 256;

  // ---- staging assignments ----
  const int arow = tid >> 2;                   // 0..63
  const int acol = tid & 3;                    // b128 chunk within 64B row
  const unsigned ldsA0 = lds_addr_of(&As[0][arow][acol * 8]);
  const unsigned ldsB0 = lds_addr_of(&Bs[0][0][0]);
  unsigned offA = ((unsigned)(mBlock + arow) * (unsigned)K + (unsigned)(acol * 8)) * 2u;
  unsigned long long gB = (unsigned long long)(uintptr_t)BT +
                          (unsigned long long)(unsigned)nBlock * (unsigned)K * 2ull;

  const v8f zero = {0.f, 0.f, 0.f, 0.f, 0.f, 0.f, 0.f, 0.f};
  v8f acc[2][4];
#pragma unroll
  for (int i = 0; i < 2; ++i)
#pragma unroll
    for (int j = 0; j < 4; ++j) acc[i][j] = zero;

  const int nk = K >> 5;

  // prologue: stage k-step 0 into buffer 0
  if (wave == 0) tdm_load_b_tile(ldsB0, gB, K);
  gB += 64;
  async_copy_b128(ldsA0, offA, A);
  offA += 64;

  for (int i = 0; i < nk; ++i) {
    if (wave == 0) wait_tensor_all();   // wave0's TDM into buf[i&1] done
    wait_async_all();                   // this wave's A-tile DMA done
    __syncthreads();                    // tiles visible; buf[(i+1)&1] free
    if (i + 1 < nk) {
      const unsigned s = (unsigned)((i + 1) & 1);
      if (wave == 0) tdm_load_b_tile(ldsB0 + (s << 14), gB, K);
      gB += 64;
      async_copy_b128(ldsA0 + (s << 12), offA, A);
      offA += 64;
    }
    const __bf16* At = &As[i & 1][0][0];
    const __bf16* Bt = &Bs[i & 1][0][0];
    v16bf aF[2], bF[4];
#pragma unroll
    for (int ii = 0; ii < 2; ++ii) aF[ii] = load_fragA(At, 32, waveM * 32 + 16 * ii, 0, lane);
#pragma unroll
    for (int j = 0; j < 4; ++j) bF[j] = load_fragB(Bt, 32, waveN * 64 + 16 * j, 0, lane);
#pragma unroll
    for (int ii = 0; ii < 2; ++ii)
#pragma unroll
      for (int j = 0; j < 4; ++j)
        acc[ii][j] = __builtin_amdgcn_wmma_f32_16x16x32_bf16(
            false, aF[ii], false, bF[j], (short)0, acc[ii][j], false, false);
  }

  const int colL = lane & 15;
  const int rOff = (lane >> 4) << 3;           // lanes>=16 hold rows M=8..15 of tile
  const int mBase = mBlock + waveM * 32;
  const int nBase = nBlock + waveN * 64;
#pragma unroll
  for (int j = 0; j < 4; ++j) {
    const int   col = nBase + 16 * j + colL;
    const float bv  = bias[col];
#pragma unroll
    for (int i = 0; i < 2; ++i) {
#pragma unroll
      for (int r = 0; r < 8; ++r) {
        int row = mBase + 16 * i + r + rOff;
        C[(size_t)row * ldc + col] = acc[i][j][r] + bv;
      }
    }
  }
}

// ---------------------------------------------------------------------------
// Fused LSTM step (one launch per t):
//  z = xz[:,t,:] + h_in @ W_h ; gates ; update c,h ; write hs[:,t,:].
//  Grid: 32 blocks (16 h-columns each) x 128 threads (4 waves = 4 gates).
//  h buffers are ping-ponged across launches (no intra-kernel cross-block race).
// ---------------------------------------------------------------------------
__launch_bounds__(128)
__global__ void lstm_step_kernel(int t,
                                 const float*  __restrict__ xz,
                                 const __bf16* __restrict__ WhT,
                                 const __bf16* __restrict__ hin,
                                 __bf16*       __restrict__ hout,
                                 float*        __restrict__ hf,
                                 float*        __restrict__ cf,
                                 __bf16*       __restrict__ hs) {
  __shared__ float zs[32][64];                 // [batch][gate*16 + hc]
  const int lane  = threadIdx.x & 31;
  const int g     = threadIdx.x >> 5;          // gate 0..3 (i,f,g,o)
  const int hBase = blockIdx.x * 16;
  const int nBase = g * H_ + hBase;            // column block in the 4H gate space

  const v8f zero = {0.f, 0.f, 0.f, 0.f, 0.f, 0.f, 0.f, 0.f};
  v8f acc[2] = {zero, zero};

  for (int k0 = 0; k0 < H_; k0 += 32) {
    v16bf bF = load_fragB(WhT, H_, nBase, k0, lane);
    v16bf a0 = load_fragA(hin, H_, 0,  k0, lane);
    v16bf a1 = load_fragA(hin, H_, 16, k0, lane);
    acc[0] = __builtin_amdgcn_wmma_f32_16x16x32_bf16(false, a0, false, bF, (short)0, acc[0], false, false);
    acc[1] = __builtin_amdgcn_wmma_f32_16x16x32_bf16(false, a1, false, bF, (short)0, acc[1], false, false);
  }

  const int colL = lane & 15;
  const int rOff = (lane >> 4) << 3;
#pragma unroll
  for (int i = 0; i < 2; ++i) {
#pragma unroll
    for (int r = 0; r < 8; ++r) {
      int b = 16 * i + r + rOff;               // batch row
      int n = nBase + colL;                    // gate-space column
      zs[b][g * 16 + colL] = acc[i][r] + xz[((size_t)b * T_ + t) * FOURH + n];
    }
  }
  __syncthreads();

#pragma unroll
  for (int p = 0; p < 4; ++p) {
    int pid = threadIdx.x + p * 128;           // 512 (b,h) pairs per block
    int b   = pid >> 4;
    int hc  = pid & 15;
    int h   = hBase + hc;
    float zi = zs[b][hc], zf = zs[b][16 + hc], zg = zs[b][32 + hc], zo = zs[b][48 + hc];
    float co = cf[b * H_ + h];
    float ig = sigmoidf_(zi);
    float fg = sigmoidf_(zf);
    float gg = tanhf(zg);
    float og = sigmoidf_(zo);
    float cn = fg * co + ig * gg;
    float hn = og * tanhf(cn);
    cf[b * H_ + h]  = cn;
    hf[b * H_ + h]  = hn;
    hout[b * H_ + h] = (__bf16)hn;
    hs[((size_t)b * T_ + t) * H_ + h] = (__bf16)hn;
  }
}

__global__ void finalize_kernel(const float* __restrict__ hf, const float* __restrict__ cf,
                                float* __restrict__ out_tail) {
  int i = blockIdx.x * blockDim.x + threadIdx.x;
  if (i >= 2 * B_ * H_) return;
  out_tail[i] = (i < B_ * H_) ? hf[i] : cf[i - B_ * H_];
}

// ---------------------------------------------------------------------------
extern "C" void kernel_launch(void* const* d_in, const int* in_sizes, int n_in,
                              void* d_out, int out_size, void* d_ws, size_t ws_size,
                              hipStream_t stream) {
  const int*   inputs = (const int*)d_in[0];
  const float* hidden = (const float*)d_in[1];
  const float* cell   = (const float*)d_in[2];
  const float* emb    = (const float*)d_in[3];
  const float* Wx     = (const float*)d_in[4];
  const float* Wh     = (const float*)d_in[5];
  const float* bias_g = (const float*)d_in[6];
  const float* Wout   = (const float*)d_in[7];
  const float* bout   = (const float*)d_in[8];
  float* out = (float*)d_out;

  // workspace carve-up (256B aligned)
  char* ws = (char*)d_ws;
  size_t off = 0;
  auto walloc = [&](size_t bytes) -> void* {
    off = (off + 255) & ~size_t(255);
    void* p = ws + off;
    off += bytes;
    return p;
  };
  __bf16* WxT = (__bf16*)walloc((size_t)FOURH * KP_E * 2);   // [2048][320]
  __bf16* WhT = (__bf16*)walloc((size_t)FOURH * H_ * 2);     // [2048][512]
  __bf16* WoT = (__bf16*)walloc((size_t)V_ * H_ * 2);        // [32000][512]
  __bf16* xb  = (__bf16*)walloc((size_t)MROWS * KP_E * 2);   // [4096][320]
  float*  xz  = (float*) walloc((size_t)MROWS * FOURH * 4);  // [4096][2048]
  __bf16* hs  = (__bf16*)walloc((size_t)MROWS * H_ * 2);     // [4096][512]
  float*  hf  = (float*) walloc((size_t)B_ * H_ * 4);
  float*  cf  = (float*) walloc((size_t)B_ * H_ * 4);
  __bf16* hb0 = (__bf16*)walloc((size_t)B_ * H_ * 2);
  __bf16* hb1 = (__bf16*)walloc((size_t)B_ * H_ * 2);

  dim3 tb(32, 8);
  // weight transposes / conversions (recomputed every call: deterministic)
  transpose_convert_kernel<<<dim3((FOURH + 31) / 32, (KP_E + 31) / 32), tb, 0, stream>>>(
      Wx, WxT, E_, FOURH, KP_E);
  transpose_convert_kernel<<<dim3((FOURH + 31) / 32, (H_ + 31) / 32), tb, 0, stream>>>(
      Wh, WhT, H_, FOURH, H_);
  transpose_convert_kernel<<<dim3((V_ + 31) / 32, (H_ + 31) / 32), tb, 0, stream>>>(
      Wout, WoT, H_, V_, H_);

  embed_kernel<<<(MROWS * KP_E + 255) / 256, 256, 0, stream>>>(inputs, emb, xb);
  init_state_kernel<<<(B_ * H_ + 255) / 256, 256, 0, stream>>>(hidden, cell, hf, cf, hb0);

  // xz = x @ W_x + b   (M=4096, N=2048, K=320)
  gemm_bf16_wmma_kernel<<<dim3(FOURH / 256, MROWS / 64), 256, 0, stream>>>(
      xb, WxT, bias_g, xz, KP_E, FOURH);

  // sequential recurrence, ping-pong bf16 h buffers
  __bf16* hbufs[2] = {hb0, hb1};
  for (int t = 0; t < T_; ++t) {
    lstm_step_kernel<<<H_ / 16, 128, 0, stream>>>(
        t, xz, WhT, hbufs[t & 1], hbufs[(t + 1) & 1], hf, cf, hs);
  }

  // logits = hs @ W_out + b_out  (M=4096, N=32000, K=512) -> d_out
  gemm_bf16_wmma_kernel<<<dim3(V_ / 256, MROWS / 64), 256, 0, stream>>>(
      hs, WoT, bout, out, H_, V_);

  // h_last, c_last appended after logits
  finalize_kernel<<<(2 * B_ * H_ + 255) / 256, 256, 0, stream>>>(
      hf, cf, out + (size_t)MROWS * V_);
}
